// TemplateSegmentAssembler_31602369364498
// MI455X (gfx1250) — matrix operand order repaired and checked by reading, not compiled, not run on hardware
//
#include <hip/hip_runtime.h>
#include <hip/hip_bf16.h>
#include <cstdint>
#include <cstddef>

// Problem dims
#define BB   8
#define NNODE 2048
#define HH   256
#define KNNK 16
#define DEG  20                 // 4 sequential + 16 knn edges per source node
#define EPB  (NNODE * DEG)      // edges per batch = 40960
#define LN_EPSF 1e-5f
#define STEPF   0.1f

typedef __attribute__((ext_vector_type(16))) __bf16 v16bf;
typedef __attribute__((ext_vector_type(8)))  __bf16 bf16x8;
typedef __attribute__((ext_vector_type(8)))  float  v8f;

// ---------- bf16 helpers (explicit bit ops: no reliance on native cvt) ----------
__device__ __forceinline__ __bf16 f2bf(float f) {
  union { float f; unsigned u; } a; a.f = f;
  unsigned r = a.u + 0x7fffu + ((a.u >> 16) & 1u);   // round-to-nearest-even
  union { unsigned short u; __bf16 b; } o; o.u = (unsigned short)(r >> 16);
  return o.b;
}
__device__ __forceinline__ float bf2f(__bf16 b) {
  union { unsigned short u; __bf16 b; } i; i.b = b;
  union { float f; unsigned u; } o; o.u = ((unsigned)i.u) << 16;
  return o.f;
}
__device__ __forceinline__ float siluf(float v) { return v / (1.f + __expf(-v)); }

// ---------- WMMA fragment loaders (row-major source, per-lane row pointer) ----------
// A (16x32 bf16): lanes 0-15 hold K {k0..k0+7, k0+16..k0+23}; lanes 16-31 hold {k0+8..15, k0+24..31}
__device__ __forceinline__ v16bf load_frag_a(const __bf16* row, int k0, int hv) {
  const int ko = k0 + (hv ? 8 : 0);
  union { v16bf v; bf16x8 h[2]; } u;
  u.h[0] = *(const bf16x8*)(row + ko);
  u.h[1] = *(const bf16x8*)(row + ko + 16);
  return u.v;
}
// B (32x16 bf16), weights stored N-major (row = output col n): lanes 0-15 K {k0..k0+7, k0+16..k0+23}; lanes 16-31 K {k0+8..15, k0+24..31}
__device__ __forceinline__ v16bf load_frag_b(const __bf16* row, int k0, int hv) {
  const int ko = k0 + (hv ? 8 : 0);
  union { v16bf v; bf16x8 h[2]; } u;
  u.h[0] = *(const bf16x8*)(row + ko);
  u.h[1] = *(const bf16x8*)(row + ko + 16);
  return u.v;
}
__device__ __forceinline__ v8f wmma_bf16(v16bf a, v16bf b, v8f c) {
  return __builtin_amdgcn_wmma_f32_16x16x32_bf16(false, a, false, b, (short)0, c, false, false);
}

// ---------- prep kernels ----------
// dst[n*K + k] = src[k*Nn + n]  (f32 -> bf16, transpose to N-major)
__global__ __launch_bounds__(256) void transpose_bf_kernel(const float* __restrict__ src,
                                                           __bf16* __restrict__ dst, int K, int Nn) {
  int i = blockIdx.x * 256 + threadIdx.x;
  if (i < K * Nn) {
    int n = i / K;
    int k = i - n * K;
    dst[i] = f2bf(src[(size_t)k * Nn + n]);
  }
}
__global__ __launch_bounds__(256) void f2bf_kernel(const float* __restrict__ s,
                                                   __bf16* __restrict__ d, size_t n) {
  size_t i = (size_t)blockIdx.x * 256 + threadIdx.x;
  if (i < n) d[i] = f2bf(s[i]);
}

// ---------- KNN + edge-list kernel: one block per (b, i) ----------
__global__ __launch_bounds__(256) void knn_kernel(const float* __restrict__ coords,
                                                  int* __restrict__ nbrIdx,
                                                  float* __restrict__ nbrW) {
  __shared__ float sD[NNODE];
  __shared__ float rv[256];
  __shared__ int   ri[256];
  __shared__ int   sK[KNNK];

  const int b = blockIdx.x >> 11;          // / 2048
  const int i = blockIdx.x & (NNODE - 1);
  const int t = threadIdx.x;
  const float INF = __builtin_inff();

  const float* xb = coords + (size_t)b * NNODE * 3;
  const float xi0 = xb[i * 3 + 0], xi1 = xb[i * 3 + 1], xi2 = xb[i * 3 + 2];

  for (int j = t; j < NNODE; j += 256) {
    float dx = xb[j * 3 + 0] - xi0;
    float dy = xb[j * 3 + 1] - xi1;
    float dz = xb[j * 3 + 2] - xi2;
    sD[j] = (j == i) ? INF : (dx * dx + dy * dy + dz * dz);
  }
  __syncthreads();

  for (int p = 0; p < KNNK; ++p) {
    float best = INF; int bi = NNODE;
    for (int j = t; j < NNODE; j += 256) {
      float v = sD[j];
      if (v < best || (v == best && j < bi)) { best = v; bi = j; }
    }
    rv[t] = best; ri[t] = bi;
    __syncthreads();
    for (int s2 = 128; s2 > 0; s2 >>= 1) {
      if (t < s2) {
        float v2 = rv[t + s2]; int i2 = ri[t + s2];
        if (v2 < rv[t] || (v2 == rv[t] && i2 < ri[t])) { rv[t] = v2; ri[t] = i2; }
      }
      __syncthreads();
    }
    if (t == 0) { int wn = ri[0]; sK[p] = wn; sD[wn] = INF; }
    __syncthreads();
  }

  if (t < DEG) {
    const int base = (b * NNODE + i) * DEG;
    int dsti; float w;
    if (t < 4) {
      const int offs[4] = {-1, 1, -2, 2};
      int dd = i + offs[t];
      bool ok = (dd >= 0) && (dd < NNODE);
      dsti = ok ? dd : i;
      w = ok ? 1.f : 0.f;
    } else {
      int dd = sK[t - 4];
      dsti = dd;
      int diff = dd - i;
      w = (diff == 1 || diff == -1 || diff == 2 || diff == -2) ? 0.f : 1.f;
    }
    nbrIdx[base + t] = dsti;
    nbrW[base + t]   = w;
  }
}

// ---------- Edge MLP (WMMA) + aggregation; 4 waves x 16 edges per workgroup ----------
__global__ __launch_bounds__(128) void edge_mlp_kernel(
    const __bf16* __restrict__ Hbf, const float* __restrict__ coords,
    const int* __restrict__ nbrIdx, const float* __restrict__ nbrW,
    const __bf16* __restrict__ W1t,  // 256 x 512 (N-major)
    const float* __restrict__ eb1, const float* __restrict__ w1c,  // eW1 row 512
    const __bf16* __restrict__ W2t,  // 256 x 256
    const float* __restrict__ eb2,
    const __bf16* __restrict__ C1t,  // 256 x 256
    const float* __restrict__ cb1,
    const float* __restrict__ cW2, const float* __restrict__ cb2,
    float* __restrict__ aggMsg, float* __restrict__ aggDelta, float* __restrict__ degOut) {
  __shared__ __align__(16) __bf16 sT1[4][16 * HH];   // layer1 out, reused for layer3 out
  __shared__ __align__(16) __bf16 sMsg[4][16 * HH];  // layer2 out (msg)

  const int wave = threadIdx.x >> 5;
  const int lane = threadIdx.x & 31;
  const int r    = lane & 15;   // tile row (edge within tile)
  const int hv   = lane >> 4;   // lane half

  const int wgPerBatch = EPB / 64;  // 640
  const int b = blockIdx.x / wgPerBatch;
  const int e = (blockIdx.x % wgPerBatch) * 64 + wave * 16 + r;
  const int s = e / DEG;
  const int slot = e - s * DEG;
  const int eIdx = (b * NNODE + s) * DEG + slot;
  const int d = nbrIdx[eIdx];
  const float w = nbrW[eIdx];

  const float* xs = coords + ((size_t)b * NNODE + s) * 3;
  const float* xd = coords + ((size_t)b * NNODE + d) * 3;
  const float rx = xs[0] - xd[0], ry = xs[1] - xd[1], rz = xs[2] - xd[2];
  const float dist2 = rx * rx + ry * ry + rz * rz;

  const __bf16* hs = Hbf + ((size_t)b * NNODE + s) * HH;
  const __bf16* hd = Hbf + ((size_t)b * NNODE + d) * HH;
  __bf16* t1  = sT1[wave];
  __bf16* msg = sMsg[wave];

  // ---- Layer 1: [h_s, h_d] @ eW1[0:512] + dist2 * eW1[512] + eb1, silu
  for (int nt = 0; nt < 16; ++nt) {
    const int n0 = nt * 16;
    const __bf16* brow = W1t + (size_t)(n0 + r) * 512;
    v8f acc = {0.f, 0.f, 0.f, 0.f, 0.f, 0.f, 0.f, 0.f};
#pragma unroll
    for (int kk = 0; kk < 256; kk += 32)
      acc = wmma_bf16(load_frag_a(hs, kk, hv), load_frag_b(brow, kk, hv), acc);
#pragma unroll
    for (int kk = 0; kk < 256; kk += 32)
      acc = wmma_bf16(load_frag_a(hd, kk, hv), load_frag_b(brow, 256 + kk, hv), acc);
    const int n = n0 + r;
    const float bias = eb1[n], wc = w1c[n];
#pragma unroll
    for (int j = 0; j < 8; ++j) {
      const int m = j + hv * 8;
      const float dm = __shfl(dist2, m, 32);
      t1[m * HH + n] = f2bf(siluf(acc[j] + bias + dm * wc));
    }
  }
  __syncthreads();

  // ---- Layer 2: t1 @ eW2 + eb2, silu -> msg
  for (int nt = 0; nt < 16; ++nt) {
    const int n0 = nt * 16;
    const __bf16* brow = W2t + (size_t)(n0 + r) * 256;
    v8f acc = {0.f, 0.f, 0.f, 0.f, 0.f, 0.f, 0.f, 0.f};
#pragma unroll
    for (int kk = 0; kk < 256; kk += 32)
      acc = wmma_bf16(load_frag_a(t1 + r * HH, kk, hv), load_frag_b(brow, kk, hv), acc);
    const int n = n0 + r;
    const float bias = eb2[n];
#pragma unroll
    for (int j = 0; j < 8; ++j) {
      const int m = j + hv * 8;
      msg[m * HH + n] = f2bf(siluf(acc[j] + bias));
    }
  }
  __syncthreads();

  // ---- Layer 3: msg @ cW1 + cb1, silu -> t2 (reuses t1 buffer)
  for (int nt = 0; nt < 16; ++nt) {
    const int n0 = nt * 16;
    const __bf16* brow = C1t + (size_t)(n0 + r) * 256;
    v8f acc = {0.f, 0.f, 0.f, 0.f, 0.f, 0.f, 0.f, 0.f};
#pragma unroll
    for (int kk = 0; kk < 256; kk += 32)
      acc = wmma_bf16(load_frag_a(msg + r * HH, kk, hv), load_frag_b(brow, kk, hv), acc);
    const int n = n0 + r;
    const float bias = cb1[n];
#pragma unroll
    for (int j = 0; j < 8; ++j) {
      const int m = j + hv * 8;
      t1[m * HH + n] = f2bf(siluf(acc[j] + bias));
    }
  }
  __syncthreads();

  // ---- Layer 4: coef = tanh(t2 . cW2 + cb2) (split across lane halves)
  float part = 0.f;
  {
    const __bf16* t2row = t1 + r * HH;
    const int kb = hv * 128;
    for (int k = 0; k < 128; ++k) part += bf2f(t2row[kb + k]) * cW2[kb + k];
  }
  part += __shfl_xor(part, 16, 32);
  const float coef = tanhf(part + cb2[0]);

  // ---- Aggregation (native f32 global atomics)
  if (hv == 0) {
    const float cw = coef * w;
    float* dl = aggDelta + ((size_t)b * NNODE + s) * 3;
    unsafeAtomicAdd(dl + 0, rx * cw);
    unsafeAtomicAdd(dl + 1, ry * cw);
    unsafeAtomicAdd(dl + 2, rz * cw);
    unsafeAtomicAdd(degOut + (size_t)b * NNODE + s, w);
  }
  for (int m = 0; m < 16; ++m) {
    const float wm = __shfl(w, m, 32);
    const int   sm = __shfl(s, m, 32);
    if (wm != 0.f) {
      float* dstp = aggMsg + ((size_t)b * NNODE + sm) * HH;
      const __bf16* mr = msg + m * HH;
#pragma unroll
      for (int c = 0; c < HH; c += 32)
        unsafeAtomicAdd(dstp + c + lane, bf2f(mr[c + lane]));
    }
  }
}

// ---------- Node MLP (WMMA): 4 waves x 16 rows per workgroup ----------
__global__ __launch_bounds__(128) void node_mlp_kernel(
    const __bf16* __restrict__ Hbf, const __bf16* __restrict__ Mbf,
    const float* __restrict__ hidden,
    const __bf16* __restrict__ N1t,  // 256 x 512
    const float* __restrict__ nb1,
    const __bf16* __restrict__ N2t,  // 256 x 256
    const float* __restrict__ nb2,
    float* __restrict__ hn) {
  __shared__ __align__(16) __bf16 sU[4][16 * HH];
  const int wave = threadIdx.x >> 5;
  const int lane = threadIdx.x & 31;
  const int r = lane & 15, hv = lane >> 4;
  const int rowBase = blockIdx.x * 64 + wave * 16;
  const __bf16* ha = Hbf + (size_t)(rowBase + r) * HH;
  const __bf16* ma = Mbf + (size_t)(rowBase + r) * HH;
  __bf16* u = sU[wave];

  for (int nt = 0; nt < 16; ++nt) {
    const int n0 = nt * 16;
    const __bf16* brow = N1t + (size_t)(n0 + r) * 512;
    v8f acc = {0.f, 0.f, 0.f, 0.f, 0.f, 0.f, 0.f, 0.f};
#pragma unroll
    for (int kk = 0; kk < 256; kk += 32)
      acc = wmma_bf16(load_frag_a(ha, kk, hv), load_frag_b(brow, kk, hv), acc);
#pragma unroll
    for (int kk = 0; kk < 256; kk += 32)
      acc = wmma_bf16(load_frag_a(ma, kk, hv), load_frag_b(brow, 256 + kk, hv), acc);
    const int n = n0 + r;
    const float bias = nb1[n];
#pragma unroll
    for (int j = 0; j < 8; ++j) u[(j + hv * 8) * HH + n] = f2bf(siluf(acc[j] + bias));
  }
  __syncthreads();

  for (int nt = 0; nt < 16; ++nt) {
    const int n0 = nt * 16;
    const __bf16* brow = N2t + (size_t)(n0 + r) * 256;
    v8f acc = {0.f, 0.f, 0.f, 0.f, 0.f, 0.f, 0.f, 0.f};
#pragma unroll
    for (int kk = 0; kk < 256; kk += 32)
      acc = wmma_bf16(load_frag_a(u + r * HH, kk, hv), load_frag_b(brow, kk, hv), acc);
    const int n = n0 + r;
    const float bias = nb2[n];
#pragma unroll
    for (int j = 0; j < 8; ++j) {
      const size_t row = rowBase + j + hv * 8;
      hn[row * HH + n] = acc[j] + bias + hidden[row * HH + n];
    }
  }
}

// ---------- LayerNorm: one block per row ----------
__global__ __launch_bounds__(256) void ln_kernel(const float* __restrict__ hn,
                                                 const float* __restrict__ g,
                                                 const float* __restrict__ bvec,
                                                 float* __restrict__ outH) {
  __shared__ float ssum[256];
  __shared__ float ssq[256];
  const int row = blockIdx.x;
  const int t = threadIdx.x;
  const float v = hn[(size_t)row * HH + t];
  ssum[t] = v; ssq[t] = v * v;
  __syncthreads();
  for (int s2 = 128; s2 > 0; s2 >>= 1) {
    if (t < s2) { ssum[t] += ssum[t + s2]; ssq[t] += ssq[t + s2]; }
    __syncthreads();
  }
  const float mu  = ssum[0] * (1.f / HH);
  const float var = ssq[0] * (1.f / HH) - mu * mu;
  outH[(size_t)row * HH + t] = (v - mu) * rsqrtf(var + LN_EPSF) * g[t] + bvec[t];
}

// ---------- Coordinate update ----------
__global__ __launch_bounds__(256) void xnew_kernel(const float* __restrict__ coords,
                                                   const float* __restrict__ aggDelta,
                                                   const float* __restrict__ deg,
                                                   float* __restrict__ outX) {
  int i = blockIdx.x * 256 + threadIdx.x;
  if (i < BB * NNODE) {
    float dg = fmaxf(deg[i], 1.f);
    float inv = STEPF / dg;
#pragma unroll
    for (int c = 0; c < 3; ++c)
      outX[(size_t)i * 3 + c] = coords[(size_t)i * 3 + c] + aggDelta[(size_t)i * 3 + c] * inv;
  }
}

extern "C" void kernel_launch(void* const* d_in, const int* in_sizes, int n_in,
                              void* d_out, int out_size, void* d_ws, size_t ws_size,
                              hipStream_t stream) {
  (void)in_sizes; (void)n_in; (void)out_size; (void)ws_size;

  const float* hidden = (const float*)d_in[0];
  const float* coords = (const float*)d_in[1];
  const float* eW1 = (const float*)d_in[3];
  const float* eb1 = (const float*)d_in[4];
  const float* eW2 = (const float*)d_in[5];
  const float* eb2 = (const float*)d_in[6];
  const float* cW1 = (const float*)d_in[7];
  const float* cb1 = (const float*)d_in[8];
  const float* cW2 = (const float*)d_in[9];
  const float* cb2 = (const float*)d_in[10];
  const float* nW1 = (const float*)d_in[11];
  const float* nb1 = (const float*)d_in[12];
  const float* nW2 = (const float*)d_in[13];
  const float* nb2 = (const float*)d_in[14];
  const float* ln_g = (const float*)d_in[15];
  const float* ln_b = (const float*)d_in[16];

  char* ws = (char*)d_ws;
  size_t off = 0;
  auto take = [&](size_t bytes) -> void* {
    void* p = ws + off;
    off += (bytes + 255) & ~(size_t)255;
    return p;
  };
  const size_t BNH = (size_t)BB * NNODE * HH;

  __bf16* Hbf = (__bf16*)take(BNH * 2);
  __bf16* Mbf = (__bf16*)take(BNH * 2);
  __bf16* W1t = (__bf16*)take((size_t)256 * 512 * 2);
  __bf16* W2t = (__bf16*)take((size_t)256 * 256 * 2);
  __bf16* C1t = (__bf16*)take((size_t)256 * 256 * 2);
  __bf16* N1t = (__bf16*)take((size_t)256 * 512 * 2);
  __bf16* N2t = (__bf16*)take((size_t)256 * 256 * 2);
  float* aggMsg   = (float*)take(BNH * 4);                      // these three stay contiguous
  float* aggDelta = (float*)take((size_t)BB * NNODE * 3 * 4);
  float* degBuf   = (float*)take((size_t)BB * NNODE * 4);
  int*   nbrIdx = (int*)take((size_t)BB * NNODE * DEG * 4);
  float* nbrW   = (float*)take((size_t)BB * NNODE * DEG * 4);
  float* hn     = (float*)take(BNH * 4);

  float* outH = (float*)d_out;
  float* outX = outH + BNH;

  // 1) weight prep (bf16, N-major transpose)
  transpose_bf_kernel<<<(512 * 256 + 255) / 256, 256, 0, stream>>>(eW1, W1t, 512, 256);
  transpose_bf_kernel<<<(256 * 256 + 255) / 256, 256, 0, stream>>>(eW2, W2t, 256, 256);
  transpose_bf_kernel<<<(256 * 256 + 255) / 256, 256, 0, stream>>>(cW1, C1t, 256, 256);
  transpose_bf_kernel<<<(512 * 256 + 255) / 256, 256, 0, stream>>>(nW1, N1t, 512, 256);
  transpose_bf_kernel<<<(256 * 256 + 255) / 256, 256, 0, stream>>>(nW2, N2t, 256, 256);
  f2bf_kernel<<<(int)((BNH + 255) / 256), 256, 0, stream>>>(hidden, Hbf, BNH);

  // 2) zero aggregation buffers (contiguous region)
  hipMemsetAsync(aggMsg, 0, BNH * 4 + (size_t)BB * NNODE * 3 * 4 + (size_t)BB * NNODE * 4, stream);

  // 3) KNN + edge list
  knn_kernel<<<BB * NNODE, 256, 0, stream>>>(coords, nbrIdx, nbrW);

  // 4) edge MLP (WMMA) + aggregation
  edge_mlp_kernel<<<BB * (EPB / 64), 128, 0, stream>>>(
      Hbf, coords, nbrIdx, nbrW,
      W1t, eb1, eW1 + (size_t)512 * 256,   // w1c = last row of eW1 (dist2 column), kept f32
      W2t, eb2, C1t, cb1, cW2, cb2,
      aggMsg, aggDelta, degBuf);

  // 5) agg_msg -> bf16 for node GEMM; coordinate output
  f2bf_kernel<<<(int)((BNH + 255) / 256), 256, 0, stream>>>(aggMsg, Mbf, BNH);
  xnew_kernel<<<(BB * NNODE + 255) / 256, 256, 0, stream>>>(coords, aggDelta, degBuf, outX);

  // 6) node MLP (WMMA), then LayerNorm
  node_mlp_kernel<<<(BB * NNODE) / 64, 128, 0, stream>>>(Hbf, Mbf, hidden, N1t, nb1, N2t, nb2, hn);
  ln_kernel<<<BB * NNODE, 256, 0, stream>>>(hn, ln_g, ln_b, outH);
}
